// GraphMixer_214748365156
// MI455X (gfx1250) — matrix operand order repaired and checked by reading, not compiled, not run on hardware
//
#include <hip/hip_runtime.h>

#define B_   1024
#define N_   128
#define H_   64
#define ID_  8
#define GIN_ 9

// Dynamic LDS layout (bytes):
//  s_A    : 128*64 f32  = 32768
//  s_B    : 128*64 f32  = 32768
//  s_attB : 128*128 bf16= 32768
//  s_WhT  : 64*128 bf16 = 16384
//  s_xg   : 128*9 f32   =  4608
//  s_ei/ej/m1/m2/g1/red =  1856
#define SMEM_BYTES 121152

typedef __attribute__((ext_vector_type(16))) __bf16        v16bf;
typedef __attribute__((ext_vector_type(8)))  float         v8f;
typedef __attribute__((ext_vector_type(4)))  unsigned int  u32x4;

union ABfrag { v16bf v; u32x4 q[2]; };

__device__ __forceinline__ unsigned short f2bf(float f) {
  unsigned int u = __float_as_uint(f);
  unsigned int r = (u + 0x7FFFu + ((u >> 16) & 1u)) >> 16;   // round-nearest-even
  return (unsigned short)r;
}
__device__ __forceinline__ float bf2f(unsigned short s) {
  return __uint_as_float(((unsigned int)s) << 16);
}

// ---- per-wave attention: 16 rows, full softmax + adj-renorm, bf16 out ----
__device__ __forceinline__ void gat_attention(const float* __restrict__ adj,
                                              const float* s_ei, const float* s_ej,
                                              unsigned short* s_attB,
                                              int wave, int lane) {
  for (int rr = 0; rr < 16; ++rr) {
    const int i = wave * 16 + rr;
    const float eiv = s_ei[i];
    float ev[4], av[4];
    float m = -3.4e38f;
    #pragma unroll
    for (int q = 0; q < 4; ++q) {
      const int j = lane + 32 * q;
      const float a = adj[i * N_ + j];
      float e = eiv + s_ej[j];
      e = (e > 0.f) ? e : (0.2f * e);            // leaky_relu(alpha=0.2)
      e = (a > 0.f) ? e : -9.0e15f;              // NEG_BIG mask
      ev[q] = e; av[q] = a;
      m = fmaxf(m, e);
    }
    #pragma unroll
    for (int off = 16; off >= 1; off >>= 1) m = fmaxf(m, __shfl_xor(m, off, 32));
    float S = 0.f, T = 0.f;
    #pragma unroll
    for (int q = 0; q < 4; ++q) {
      const float p = __expf(ev[q] - m);
      const float t = p * av[q];
      S += p; T += t; ev[q] = t;
    }
    #pragma unroll
    for (int off = 16; off >= 1; off >>= 1) {
      S += __shfl_xor(S, off, 32);
      T += __shfl_xor(T, off, 32);
    }
    // att = (exp/S)*adj / (sum+1e-8)  ==  exp*adj / (T + 1e-8*S)
    const float inv = 1.f / (T + 1e-8f * S);
    #pragma unroll
    for (int q = 0; q < 4; ++q)
      s_attB[i * N_ + lane + 32 * q] = f2bf(ev[q] * inv);
  }
}

// ---- h_prime(128x64) = att(128x128,bf16) @ Wh(128x64,bf16 stored transposed) ----
// v_wmma_f32_16x16x32_bf16; wave w owns M-strip m0=16w.
// K-outer / N-inner: each A fragment loaded once, reused for 4 WMMAs (4 live accs).
__device__ __forceinline__ void gat_spmm(const unsigned short* s_attB,
                                         const unsigned short* s_WhT,
                                         float* s_out, int wave, int lane) {
  const int m0     = wave * 16;
  const int mrow   = m0 + (lane & 15);
  const int hi     = (lane >> 4) & 1;
  const int khalfA = hi * 16;    // A: upper half-wave holds K+16
  const int khalfB = hi * 16;    // B: upper half-wave holds K+16
  const int rbase  = m0 + hi * 8;
  const int ncol   = lane & 15;

  v8f acc[4] = {{}, {}, {}, {}};
  #pragma unroll
  for (int kt = 0; kt < 4; ++kt) {
    const int k0 = kt * 32;
    ABfrag a;
    a.q[0] = *(const u32x4*)(s_attB + mrow * N_ + k0 + khalfA);        // K kb..kb+7
    a.q[1] = *(const u32x4*)(s_attB + mrow * N_ + k0 + khalfA + 8);    // K kb+8..kb+15
    #pragma unroll
    for (int nt = 0; nt < 4; ++nt) {
      ABfrag b;
      const u32x4* pb = (const u32x4*)(s_WhT + (nt * 16 + ncol) * N_ + k0 + khalfB);
      b.q[0] = pb[0]; b.q[1] = pb[1];                                  // 16 consecutive K
      acc[nt] = __builtin_amdgcn_wmma_f32_16x16x32_bf16(
                  false, a.v, false, b.v, (short)0, acc[nt], false, false);
    }
  }
  #pragma unroll
  for (int nt = 0; nt < 4; ++nt) {
    const int ocol = nt * 16 + ncol;
    #pragma unroll
    for (int v = 0; v < 8; ++v)
      s_out[(rbase + v) * H_ + ocol] = acc[nt][v];
  }
}

__global__ __launch_bounds__(256)
void graphmixer_kernel(const float* __restrict__ cvals, const float* __restrict__ adj,
                       const float* __restrict__ aemb,
                       const float* __restrict__ W1,  const float* __restrict__ a1,
                       const float* __restrict__ W2,  const float* __restrict__ a2,
                       const float* __restrict__ ln1g, const float* __restrict__ ln1b,
                       const float* __restrict__ ln2g, const float* __restrict__ ln2b,
                       const float* __restrict__ hgw, const float* __restrict__ hgb,
                       const float* __restrict__ m1w, const float* __restrict__ m1b,
                       const float* __restrict__ m2w, const float* __restrict__ m2b,
                       const float* __restrict__ hmw, const float* __restrict__ hmb,
                       const float* __restrict__ g1w, const float* __restrict__ g1b,
                       const float* __restrict__ g2w, const float* __restrict__ g2b,
                       float* __restrict__ out)
{
  extern __shared__ char smem[];
  float*          s_A    = (float*)smem;                        // h_prime1 -> h1
  float*          s_B    = s_A + N_ * H_;                       // h_prime2 -> h
  unsigned short* s_attB = (unsigned short*)(s_B + N_ * H_);    // att, bf16
  unsigned short* s_WhT  = s_attB + N_ * N_;                    // Wh^T, bf16 [o][i]
  float*          s_xg   = (float*)(s_WhT + H_ * N_);
  float*          s_ei   = s_xg + N_ * GIN_;
  float*          s_ej   = s_ei + N_;
  float*          s_m1   = s_ej + N_;
  float*          s_m2   = s_m1 + H_;
  float*          s_g1   = s_m2 + H_;
  float*          s_red  = s_g1 + H_;

  const int b    = blockIdx.x;
  const int tid  = threadIdx.x;
  const int wave = tid >> 5;
  const int lane = tid & 31;
  const float* crow = cvals + b * N_;

  __builtin_prefetch(adj + tid * 64, 0, 0);   // pull adj toward L2/L0

  // ---- x_g = concat(c_values[:,None], agent_emb) ----
  for (int idx = tid; idx < N_ * GIN_; idx += 256) {
    const int i = idx / GIN_, f = idx - i * GIN_;
    s_xg[idx] = (f == 0) ? crow[i] : aemb[i * ID_ + (f - 1)];
  }
  __syncthreads();

  // ==================== GAT layer 1 ====================
  for (int idx = tid; idx < N_ * H_; idx += 256) {        // Wh1^T in bf16
    const int o = idx & (H_ - 1), i = idx >> 6;
    float acc = 0.f;
    #pragma unroll
    for (int f = 0; f < GIN_; ++f) acc += s_xg[i * GIN_ + f] * W1[f * H_ + o];
    s_WhT[o * N_ + i] = f2bf(acc);
  }
  __syncthreads();
  if (tid < N_) {                                          // ei = Wh@a_i, ej = Wh@a_j
    float e0 = 0.f, e1 = 0.f;
    for (int o = 0; o < H_; ++o) {
      const float w = bf2f(s_WhT[o * N_ + tid]);
      e0 += w * a1[o]; e1 += w * a1[H_ + o];
    }
    s_ei[tid] = e0; s_ej[tid] = e1;
  }
  __syncthreads();
  gat_attention(adj, s_ei, s_ej, s_attB, wave, lane);
  __syncthreads();
  gat_spmm(s_attB, s_WhT, s_A, wave, lane);                // WMMA: att @ Wh1
  __syncthreads();
  if (tid < N_) {                                          // ELU + LayerNorm1 (in place)
    float* row = s_A + tid * H_;
    float s = 0.f, s2 = 0.f;
    for (int o = 0; o < H_; ++o) {
      float v = row[o];
      v = (v > 0.f) ? v : (__expf(v) - 1.f);
      row[o] = v; s += v; s2 += v * v;
    }
    const float mu  = s * (1.f / H_);
    const float inv = rsqrtf(s2 * (1.f / H_) - mu * mu + 1e-5f);
    for (int o = 0; o < H_; ++o)
      row[o] = (row[o] - mu) * inv * ln1g[o] + ln1b[o];
  }
  __syncthreads();

  // ==================== GAT layer 2 ====================
  for (int idx = tid; idx < N_ * H_; idx += 256) {        // Wh2^T = (h1@W2)^T in bf16
    const int o = idx & (H_ - 1), i = idx >> 6;
    float acc = 0.f;
    for (int f = 0; f < H_; ++f) acc += s_A[i * H_ + f] * W2[f * H_ + o];
    s_WhT[o * N_ + i] = f2bf(acc);
  }
  __syncthreads();
  if (tid < N_) {
    float e0 = 0.f, e1 = 0.f;
    for (int o = 0; o < H_; ++o) {
      const float w = bf2f(s_WhT[o * N_ + tid]);
      e0 += w * a2[o]; e1 += w * a2[H_ + o];
    }
    s_ei[tid] = e0; s_ej[tid] = e1;
  }
  __syncthreads();
  gat_attention(adj, s_ei, s_ej, s_attB, wave, lane);
  __syncthreads();
  gat_spmm(s_attB, s_WhT, s_B, wave, lane);                // WMMA: att2 @ Wh2
  __syncthreads();
  if (tid < N_) {                                          // relu(h2+h1) + LayerNorm2
    float* r1 = s_A + tid * H_;
    float* r2 = s_B + tid * H_;
    float s = 0.f, s2 = 0.f;
    for (int o = 0; o < H_; ++o) {
      float v = r1[o] + r2[o];
      v = (v > 0.f) ? v : 0.f;
      r2[o] = v; s += v; s2 += v * v;
    }
    const float mu  = s * (1.f / H_);
    const float inv = rsqrtf(s2 * (1.f / H_) - mu * mu + 1e-5f);
    for (int o = 0; o < H_; ++o)
      r2[o] = (r2[o] - mu) * inv * ln2g[o] + ln2b[o];
  }
  __syncthreads();

  // ---- q_gat = mean_i(h[i]) . hgw + hgb  (block reduction) ----
  {
    float part = 0.f;
    for (int idx = tid; idx < N_ * H_; idx += 256)
      part += s_B[idx] * hgw[idx & (H_ - 1)];
    #pragma unroll
    for (int off = 16; off >= 1; off >>= 1) part += __shfl_xor(part, off, 32);
    if (lane == 0) s_red[wave] = part;
  }
  __syncthreads();

  // ---- MLP + gate side paths ----
  if (tid < H_) {
    float acc = m1b[tid], g = g1b[tid];
    for (int j = 0; j < N_; ++j) {
      acc += crow[j] * m1w[j * H_ + tid];
      g   += crow[j] * g1w[j * H_ + tid];
    }
    s_m1[tid] = (acc > 0.f) ? acc : 0.f;
    s_g1[tid] = (g   > 0.f) ? g   : 0.f;
  }
  __syncthreads();
  if (tid < H_) {
    float acc = m2b[tid];
    for (int j = 0; j < H_; ++j) acc += s_m1[j] * m2w[j * H_ + tid];
    s_m2[tid] = (acc > 0.f) ? acc : 0.f;
  }
  __syncthreads();
  if (tid == 0) {
    float tot = 0.f;
    for (int w = 0; w < 8; ++w) tot += s_red[w];
    const float q_gat = tot * (1.f / N_) + hgb[0];
    float q_mlp = hmb[0], gsc = g2b[0];
    for (int o = 0; o < H_; ++o) { q_mlp += s_m2[o] * hmw[o]; gsc += s_g1[o] * g2w[o]; }
    const float gate = 1.f / (1.f + __expf(-gsc));
    out[b] = q_mlp + gate * q_gat;                         // GAT_SCALE = 1.0
  }
}

extern "C" void kernel_launch(void* const* d_in, const int* in_sizes, int n_in,
                              void* d_out, int out_size, void* d_ws, size_t ws_size,
                              hipStream_t stream) {
  (void)in_sizes; (void)n_in; (void)out_size; (void)d_ws; (void)ws_size;
  const float* cvals = (const float*)d_in[0];
  const float* adj   = (const float*)d_in[1];
  const float* aemb  = (const float*)d_in[2];
  const float* W1    = (const float*)d_in[3];
  const float* a1    = (const float*)d_in[4];
  const float* W2    = (const float*)d_in[5];
  const float* a2    = (const float*)d_in[6];
  const float* ln1g  = (const float*)d_in[7];
  const float* ln1b  = (const float*)d_in[8];
  const float* ln2g  = (const float*)d_in[9];
  const float* ln2b  = (const float*)d_in[10];
  const float* hgw   = (const float*)d_in[11];
  const float* hgb   = (const float*)d_in[12];
  const float* m1w   = (const float*)d_in[13];
  const float* m1b   = (const float*)d_in[14];
  const float* m2w   = (const float*)d_in[15];
  const float* m2b   = (const float*)d_in[16];
  const float* hmw   = (const float*)d_in[17];
  const float* hmb   = (const float*)d_in[18];
  const float* g1w   = (const float*)d_in[19];
  const float* g1b   = (const float*)d_in[20];
  const float* g2w   = (const float*)d_in[21];
  const float* g2b   = (const float*)d_in[22];
  float* out = (float*)d_out;

  // 118 KB dynamic LDS per workgroup (CDNA5 WGP has 320 KB).
  hipFuncSetAttribute((const void*)graphmixer_kernel,
                      hipFuncAttributeMaxDynamicSharedMemorySize, SMEM_BYTES);

  graphmixer_kernel<<<dim3(B_), dim3(256), SMEM_BYTES, stream>>>(
      cvals, adj, aemb, W1, a1, W2, a2, ln1g, ln1b, ln2g, ln2b,
      hgw, hgb, m1w, m1b, m2w, m2b, hmw, hmb, g1w, g1b, g2w, g2b, out);
}
